// AttnTransliterator_11149735100543
// MI455X (gfx1250) — matrix operand order repaired and checked by reading, not compiled, not run on hardware
//
#include <hip/hip_runtime.h>
#include <hip/hip_bf16.h>

typedef __bf16 bf16_t;
typedef __attribute__((ext_vector_type(16))) __bf16 v16bf;
typedef __attribute__((ext_vector_type(8)))  float  v8f;

static const int kB = 1024, kS = 32, kT = 32;
static const int kE = 128, kHe = 256, kHd = 256, kA = 256, kVt = 256;

// ---------------------------------------------------------------------------
// Blocked bf16 WMMA GEMM:  Cf[M,N] = A[M,K](bf16) @ W[N,K](bf16)^T + bias
//
// Block = 256 threads (8 waves) -> 128x64 C tile. Waves stack along M, so all
// 8 waves share each 64x32 weight chunk, staged once per block into LDS and
// double-buffered. Per iteration: global-prefetch next B chunk + next A
// fragment into registers, ds_load all 4 B fragments (single clause/wait),
// then 4 back-to-back WMMAs, then ds_store the prefetched chunk + barrier.
// B rows padded to 40 halves to avoid LDS bank conflicts.
//
// Fragment packing follows CDNA5 ISA 7.12.2:
//   A (16x32 bf16): lane l -> row m=l%16; halves j: k = (j/8)*16 + (l/16)*8 + j%8
//   B (32x16 bf16): lane l -> col n=l%16; halves j: k = (j/8)*16 + (l/16)*8 + j%8
//   C (16x16 f32):  vgpr r, lane l -> m = r + 8*(l/16), n = l%16
// ---------------------------------------------------------------------------
union FragBF16 { v16bf v; uint4 q[2]; };

#define BSTRIDE 40   // halves per B row in LDS (32 data + 8 pad)

__global__ __launch_bounds__(256) void gemm_bf16_wmma(
    const bf16_t* __restrict__ A, int lda,
    const bf16_t* __restrict__ W, int ldb,
    const float*  __restrict__ bias,
    float*  __restrict__ Cf, long ldc,
    bf16_t* __restrict__ Ch, long ldch,
    int M, int N, int K)
{
    const int stripsN = N >> 6;
    const int bm = blockIdx.x / stripsN;       // 128-row block
    const int bn = blockIdx.x % stripsN;       // 64-col strip
    const int wid  = threadIdx.x >> 5;         // wave -> M sub-tile 0..7
    const int lane = threadIdx.x & 31;
    const int g  = lane >> 4;                  // half-wave group
    const int ln = lane & 15;

    __shared__ bf16_t Bs[2][64 * BSTRIDE];

    // ---- B staging assignment: thread t loads 8 halves of one weight row ---
    const int tr = threadIdx.x >> 2;           // 0..63 : col within strip
    const int tc = (threadIdx.x & 3) * 8;      // 0,8,16,24 : k-offset
    const bf16_t* Wrow = W + (long)(bn * 64 + tr) * ldb + tc;
    bf16_t* BsStore0 = &Bs[0][tr * BSTRIDE + tc];
    bf16_t* BsStore1 = &Bs[1][tr * BSTRIDE + tc];

    const long rowA = (long)(bm * 128 + wid * 16 + ln) * lda;
    const bf16_t* Arow = A + rowA + g * 8;

    v8f acc[4];
#pragma unroll
    for (int j = 0; j < 4; ++j)
        acc[j] = (v8f){0.f,0.f,0.f,0.f,0.f,0.f,0.f,0.f};

    // prologue: stage k0 = 0 B chunk into LDS, k0 = 0 A fragment into regs
    *reinterpret_cast<uint4*>(BsStore0) = *reinterpret_cast<const uint4*>(Wrow);
    FragBF16 afc;
    afc.q[0] = *reinterpret_cast<const uint4*>(Arow);
    afc.q[1] = *reinterpret_cast<const uint4*>(Arow + 16);
    __syncthreads();

    const int nk = K >> 5;
    for (int i = 0; i < nk; ++i) {
        const int k0 = i << 5;
        // global prefetch of next chunk (B -> regs for LDS, A -> regs)
        uint4 bnext;
        FragBF16 afn;
        if (i + 1 < nk) {
            bnext    = *reinterpret_cast<const uint4*>(Wrow + k0 + 32);
            afn.q[0] = *reinterpret_cast<const uint4*>(Arow + k0 + 32);
            afn.q[1] = *reinterpret_cast<const uint4*>(Arow + k0 + 48);
        }

        // stage all 4 B fragments (one ds clause, one wait), then 4 WMMAs
        const bf16_t* bsBuf = Bs[i & 1];
        FragBF16 bf[4];
#pragma unroll
        for (int j = 0; j < 4; ++j) {
            const bf16_t* pb = bsBuf + (j * 16 + ln) * BSTRIDE + g * 8;
            bf[j].q[0] = *reinterpret_cast<const uint4*>(pb);
            bf[j].q[1] = *reinterpret_cast<const uint4*>(pb + 16);
        }
#pragma unroll
        for (int j = 0; j < 4; ++j)
            acc[j] = __builtin_amdgcn_wmma_f32_16x16x32_bf16(
                false, afc.v, false, bf[j].v, (short)0, acc[j], false, false);

        if (i + 1 < nk) {
            *reinterpret_cast<uint4*>((i & 1) ? BsStore0 : BsStore1) = bnext;
            afc = afn;
        }
        __syncthreads();
    }

#pragma unroll
    for (int j = 0; j < 4; ++j) {
        const int col = bn * 64 + j * 16 + ln;
        const float bv = bias ? bias[col] : 0.f;
#pragma unroll
        for (int r = 0; r < 8; ++r) {
            const int m = bm * 128 + wid * 16 + r + 8 * g;
            const float v = acc[j][r] + bv;
            Cf[(long)m * ldc + col] = v;
            if (Ch) Ch[(long)m * ldch + col] = (bf16_t)v;
        }
    }
}

// ---------------------------------------------------------------------------
// Elementwise / gather helpers
// ---------------------------------------------------------------------------
__global__ void convert_to_bf16(const float* __restrict__ src, int src_ld, int src_off,
                                bf16_t* __restrict__ dst, int rows, int cols)
{
    int i = blockIdx.x * blockDim.x + threadIdx.x;
    if (i >= rows * cols) return;
    int r = i / cols, c = i % cols;
    dst[i] = (bf16_t)src[(long)r * src_ld + src_off + c];
}

__global__ void zero_f32(float* p, int n)
{ int i = blockIdx.x * blockDim.x + threadIdx.x; if (i < n) p[i] = 0.f; }

__global__ void zero_bf16(bf16_t* p, int n)
{ int i = blockIdx.x * blockDim.x + threadIdx.x; if (i < n) p[i] = (bf16_t)0.f; }

// outputs[:, 0, :] = 0
__global__ void zero_out_t0(float* out)
{
    int i = blockIdx.x * blockDim.x + threadIdx.x;
    if (i >= kB * kVt) return;
    out[(long)(i / kVt) * kT * kVt + (i % kVt)] = 0.f;
}

// x_all[s][b][e] = enc_emb[src[b,s]][e]   (bf16)
__global__ void embed_enc(const int* __restrict__ src, const float* __restrict__ emb,
                          bf16_t* __restrict__ x)
{
    int i = blockIdx.x * blockDim.x + threadIdx.x;
    if (i >= kS * kB * kE) return;
    int s = i / (kB * kE);
    int rem = i % (kB * kE);
    int b = rem / kE, e = rem % kE;
    int tok = src[b * kS + s];
    x[i] = (bf16_t)emb[tok * kE + e];
}

// xcat[b][0..127] = dec_emb[trg[b,t]][e]   (bf16)
__global__ void embed_dec_step(const int* __restrict__ trg, const float* __restrict__ emb,
                               bf16_t* __restrict__ xcat, int t)
{
    int i = blockIdx.x * blockDim.x + threadIdx.x;
    if (i >= kB * kE) return;
    int b = i / kE, e = i % kE;
    int tok = trg[b * kT + t];
    xcat[(long)b * (kE + 2 * kHe) + e] = (bf16_t)emb[tok * kE + e];
}

// GRU gate fusion: h = (1-z)*n + z*h_prev, dual-precision outputs + optional extras
__global__ void gru_gate(const float* __restrict__ gi, const float* __restrict__ gh,
                         const float* __restrict__ hprev, int ldhp,
                         float* __restrict__ hf, int ldhf,
                         bf16_t* __restrict__ hh, int ldhh,
                         float* __restrict__ ef, long ldef,
                         bf16_t* __restrict__ eh, long ldeh,
                         int H)
{
    int i = blockIdx.x * blockDim.x + threadIdx.x;
    if (i >= kB * H) return;
    int b = i / H, j = i % H;
    const long gbase = (long)b * 3 * H + j;
    float ir = gi[gbase], iz = gi[gbase + H], in = gi[gbase + 2 * H];
    float hr = gh[gbase], hz = gh[gbase + H], hn = gh[gbase + 2 * H];
    float hp = hprev[(long)b * ldhp + j];
    float r = 1.f / (1.f + __expf(-(ir + hr)));
    float z = 1.f / (1.f + __expf(-(iz + hz)));
    float n = tanhf(in + r * hn);
    float h = (1.f - z) * n + z * hp;
    hf[(long)b * ldhf + j] = h;
    hh[(long)b * ldhh + j] = (bf16_t)h;
    if (ef) ef[(long)b * ldef + j] = h;
    if (eh) eh[(long)b * ldeh + j] = (bf16_t)h;
}

// Fused attention step (one block per batch row):
// scores[s] = v . tanh(enc_pre[b,s,:] + q[b,:]); softmax over s; ctx = aw @ enc_out
// ctx written (bf16) into xcat[:,128:640] and hcat[:,256:768]
__global__ __launch_bounds__(256) void attn_step(
    const float* __restrict__ enc_pre, const float* __restrict__ q,
    const float* __restrict__ v_attn,  const float* __restrict__ enc_out_f,
    bf16_t* __restrict__ xcat, bf16_t* __restrict__ hcat)
{
    const int b = blockIdx.x, tid = threadIdx.x;
    __shared__ float part[256];
    __shared__ float sc[kS];
    __shared__ float aw[kS];

    const int s = tid >> 3, pp = tid & 7;           // 32 s-rows x 8 partials
    const float* ep = enc_pre + ((long)b * kS + s) * kA + pp * 32;
    const float* qq = q + (long)b * kA + pp * 32;
    const float* vv = v_attn + pp * 32;
    float acc = 0.f;
#pragma unroll 4
    for (int a = 0; a < 32; ++a) acc += vv[a] * tanhf(ep[a] + qq[a]);
    part[tid] = acc;
    __syncthreads();
    if (pp == 0) {
        float t = 0.f;
        for (int k = 0; k < 8; ++k) t += part[(s << 3) + k];
        sc[s] = t;
    }
    __syncthreads();
    if (tid == 0) {
        float mx = sc[0];
        for (int k = 1; k < kS; ++k) mx = fmaxf(mx, sc[k]);
        float sum = 0.f;
        for (int k = 0; k < kS; ++k) { float e = __expf(sc[k] - mx); aw[k] = e; sum += e; }
        float inv = 1.f / sum;
        for (int k = 0; k < kS; ++k) aw[k] *= inv;
    }
    __syncthreads();
    for (int d = tid; d < 2 * kHe; d += 256) {
        const float* eo = enc_out_f + (long)b * kS * (2 * kHe) + d;
        float c = 0.f;
#pragma unroll 8
        for (int k = 0; k < kS; ++k) c += aw[k] * eo[k * (2 * kHe)];
        bf16_t ch = (bf16_t)c;
        xcat[(long)b * (kE + 2 * kHe) + kE + d] = ch;
        hcat[(long)b * (kHd + 2 * kHe) + kHd + d] = ch;
    }
}

// ---------------------------------------------------------------------------
extern "C" void kernel_launch(void* const* d_in, const int* in_sizes, int n_in,
                              void* d_out, int out_size, void* d_ws, size_t ws_size,
                              hipStream_t stream)
{
    (void)in_sizes; (void)n_in; (void)out_size; (void)ws_size;
    const int*   src     = (const int*)  d_in[0];
    const int*   trg     = (const int*)  d_in[1];
    const float* enc_emb = (const float*)d_in[2];
    const float* eWih_f  = (const float*)d_in[3];
    const float* eWhh_f  = (const float*)d_in[4];
    const float* ebih_f  = (const float*)d_in[5];
    const float* ebhh_f  = (const float*)d_in[6];
    const float* eWih_b  = (const float*)d_in[7];
    const float* eWhh_b  = (const float*)d_in[8];
    const float* ebih_b  = (const float*)d_in[9];
    const float* ebhh_b  = (const float*)d_in[10];
    const float* Wproj   = (const float*)d_in[11];
    const float* bproj   = (const float*)d_in[12];
    const float* dec_emb = (const float*)d_in[13];
    const float* Wattn   = (const float*)d_in[14];
    const float* battn   = (const float*)d_in[15];
    const float* v_attn  = (const float*)d_in[16];
    const float* dWih    = (const float*)d_in[17];
    const float* dWhh    = (const float*)d_in[18];
    const float* dbih    = (const float*)d_in[19];
    const float* dbhh    = (const float*)d_in[20];
    const float* Wfc     = (const float*)d_in[21];
    const float* bfc     = (const float*)d_in[22];
    float* out = (float*)d_out;

    // ---- bump allocator over workspace -------------------------------------
    char* p = (char*)d_ws;
    auto alloc = [&](size_t bytes) -> void* {
        void* r = (void*)p;
        p += (bytes + 255) & ~(size_t)255;
        return r;
    };
    const int H3 = 3 * kHe;                      // 768
    bf16_t* eWihF_h = (bf16_t*)alloc((size_t)H3 * kE * 2);
    bf16_t* eWhhF_h = (bf16_t*)alloc((size_t)H3 * kHe * 2);
    bf16_t* eWihB_h = (bf16_t*)alloc((size_t)H3 * kE * 2);
    bf16_t* eWhhB_h = (bf16_t*)alloc((size_t)H3 * kHe * 2);
    bf16_t* Wproj_h = (bf16_t*)alloc((size_t)kHd * 2 * kHe * 2);
    bf16_t* Wd_h    = (bf16_t*)alloc((size_t)kA * kHd * 2);
    bf16_t* We_h    = (bf16_t*)alloc((size_t)kA * 2 * kHe * 2);
    bf16_t* dWih_h  = (bf16_t*)alloc((size_t)H3 * (kE + 2 * kHe) * 2);
    bf16_t* dWhh_h  = (bf16_t*)alloc((size_t)H3 * kHd * 2);
    bf16_t* Wfc_h   = (bf16_t*)alloc((size_t)kVt * (kHd + 2 * kHe) * 2);
    bf16_t* xAll_h  = (bf16_t*)alloc((size_t)kS * kB * kE * 2);
    bf16_t* henc_h  = (bf16_t*)alloc((size_t)kB * 2 * kHe * 2);
    bf16_t* encOut_h= (bf16_t*)alloc((size_t)kB * kS * 2 * kHe * 2);
    bf16_t* hd_h    = (bf16_t*)alloc((size_t)kB * kHd * 2);
    bf16_t* xcat_h  = (bf16_t*)alloc((size_t)kB * (kE + 2 * kHe) * 2);
    bf16_t* hcat_h  = (bf16_t*)alloc((size_t)kB * (kHd + 2 * kHe) * 2);
    float*  henc_f  = (float*) alloc((size_t)kB * 2 * kHe * 4);
    float*  encOut_f= (float*) alloc((size_t)kB * kS * 2 * kHe * 4);
    float*  encPre_f= (float*) alloc((size_t)kB * kS * kA * 4);
    float*  gi_f    = (float*) alloc((size_t)kB * H3 * 4);
    float*  gh_f    = (float*) alloc((size_t)kB * H3 * 4);
    float*  hd_f    = (float*) alloc((size_t)kB * kHd * 4);
    float*  q_f     = (float*) alloc((size_t)kB * kA * 4);

    auto conv = [&](const float* s, int ld, int off, bf16_t* dmat, int rows, int cols) {
        int n = rows * cols;
        convert_to_bf16<<<(n + 255) / 256, 256, 0, stream>>>(s, ld, off, dmat, rows, cols);
    };
    auto gemm = [&](const bf16_t* A, int lda, const bf16_t* W, int ldb, const float* bias,
                    float* Cf, long ldc, bf16_t* Ch, long ldch, int M, int N, int K) {
        int blocks = (M / 128) * (N / 64);
        gemm_bf16_wmma<<<blocks, 256, 0, stream>>>(A, lda, W, ldb, bias,
                                                   Cf, ldc, Ch, ldch, M, N, K);
    };

    // ---- weight conversion (once per call, deterministic) ------------------
    conv(eWih_f, kE, 0, eWihF_h, H3, kE);
    conv(eWhh_f, kHe, 0, eWhhF_h, H3, kHe);
    conv(eWih_b, kE, 0, eWihB_h, H3, kE);
    conv(eWhh_b, kHe, 0, eWhhB_h, H3, kHe);
    conv(Wproj, 2 * kHe, 0, Wproj_h, kHd, 2 * kHe);
    conv(Wattn, kHd + 2 * kHe, 0,   Wd_h, kA, kHd);
    conv(Wattn, kHd + 2 * kHe, kHd, We_h, kA, 2 * kHe);
    conv(dWih, kE + 2 * kHe, 0, dWih_h, H3, kE + 2 * kHe);
    conv(dWhh, kHd, 0, dWhh_h, H3, kHd);
    conv(Wfc, kHd + 2 * kHe, 0, Wfc_h, kVt, kHd + 2 * kHe);

    // ---- encoder -----------------------------------------------------------
    {
        int n = kS * kB * kE;
        embed_enc<<<(n + 255) / 256, 256, 0, stream>>>(src, enc_emb, xAll_h);
    }
    {
        int n = kB * 2 * kHe;
        zero_f32 <<<(n + 255) / 256, 256, 0, stream>>>(henc_f, n);
        zero_bf16<<<(n + 255) / 256, 256, 0, stream>>>(henc_h, n);
    }
    const int gateBlocks = (kB * kHe + 255) / 256;
    const long eoLd = (long)kS * 2 * kHe;      // enc_out batch stride (row = b)
    // forward direction
    for (int s = 0; s < kS; ++s) {
        gemm(xAll_h + (size_t)s * kB * kE, kE, eWihF_h, kE, ebih_f,
             gi_f, H3, nullptr, 0, kB, H3, kE);
        gemm(henc_h, 2 * kHe, eWhhF_h, kHe, ebhh_f,
             gh_f, H3, nullptr, 0, kB, H3, kHe);
        gru_gate<<<gateBlocks, 256, 0, stream>>>(
            gi_f, gh_f, henc_f, 2 * kHe,
            henc_f, 2 * kHe, henc_h, 2 * kHe,
            encOut_f + (long)s * 2 * kHe, eoLd,
            encOut_h + (long)s * 2 * kHe, eoLd, kHe);
    }
    // backward direction
    for (int s = kS - 1; s >= 0; --s) {
        gemm(xAll_h + (size_t)s * kB * kE, kE, eWihB_h, kE, ebih_b,
             gi_f, H3, nullptr, 0, kB, H3, kE);
        gemm(henc_h + kHe, 2 * kHe, eWhhB_h, kHe, ebhh_b,
             gh_f, H3, nullptr, 0, kB, H3, kHe);
        gru_gate<<<gateBlocks, 256, 0, stream>>>(
            gi_f, gh_f, henc_f + kHe, 2 * kHe,
            henc_f + kHe, 2 * kHe, henc_h + kHe, 2 * kHe,
            encOut_f + (long)s * 2 * kHe + kHe, eoLd,
            encOut_h + (long)s * 2 * kHe + kHe, eoLd, kHe);
    }
    // hdec = [hf,hb] @ Wproj^T + bproj  (fp32 + bf16)
    gemm(henc_h, 2 * kHe, Wproj_h, 2 * kHe, bproj,
         hd_f, kHd, hd_h, kHd, kB, kHd, 2 * kHe);
    // enc_pre = enc_out @ We^T + battn  (M = B*S)
    gemm(encOut_h, 2 * kHe, We_h, 2 * kHe, battn,
         encPre_f, kA, nullptr, 0, kB * kS, kA, 2 * kHe);

    // ---- decoder -----------------------------------------------------------
    zero_out_t0<<<(kB * kVt + 255) / 256, 256, 0, stream>>>(out);
    for (int t = 0; t < kT - 1; ++t) {
        embed_dec_step<<<(kB * kE + 255) / 256, 256, 0, stream>>>(trg, dec_emb, xcat_h, t);
        // q = h @ Wd^T
        gemm(hd_h, kHd, Wd_h, kHd, nullptr, q_f, kA, nullptr, 0, kB, kA, kHd);
        // attention: scores -> softmax -> ctx (into xcat/hcat, bf16)
        attn_step<<<kB, 256, 0, stream>>>(encPre_f, q_f, v_attn, encOut_f, xcat_h, hcat_h);
        // gi = [emb,ctx] @ dWih^T + dbih ; gh = h @ dWhh^T + dbhh
        gemm(xcat_h, kE + 2 * kHe, dWih_h, kE + 2 * kHe, dbih,
             gi_f, H3, nullptr, 0, kB, H3, kE + 2 * kHe);
        gemm(hd_h, kHd, dWhh_h, kHd, dbhh, gh_f, H3, nullptr, 0, kB, H3, kHd);
        gru_gate<<<gateBlocks, 256, 0, stream>>>(
            gi_f, gh_f, hd_f, kHd,
            hd_f, kHd, hd_h, kHd,
            nullptr, 0, hcat_h, kHd + 2 * kHe, kHd);
        // logits = [hn,ctx] @ Wfc^T + bfc  -> out[:, t+1, :]
        gemm(hcat_h, kHd + 2 * kHe, Wfc_h, kHd + 2 * kHe, bfc,
             out + (long)(t + 1) * kVt, (long)kT * kVt, nullptr, 0,
             kB, kVt, kHd + 2 * kHe);
    }
}
